// BigBirdAttentionLayer_4578435137975
// MI455X (gfx1250) — compile-verified, matrix operand order checked by reading, not run on hardware
//
#include <hip/hip_runtime.h>
#include <math.h>
#include <stdint.h>

// Problem constants (from reference)
#define B_  2
#define S_  4096
#define DM  768
#define H_  12
#define FF  3072
#define NB  64
#define M_  (B_ * S_)   // 8192 rows

typedef __attribute__((ext_vector_type(16))) __bf16       v16bf;
typedef __attribute__((ext_vector_type(8)))  float        v8f;
typedef __attribute__((ext_vector_type(4)))  unsigned int u32x4;

union Frag {
    v16bf v;
    u32x4 q[2];
    unsigned short s[16];
};

__device__ __forceinline__ unsigned short f2bf(float f) {
    unsigned int u = __float_as_uint(f);
    u += 0x7fffu + ((u >> 16) & 1u);      // round-to-nearest-even
    return (unsigned short)(u >> 16);
}

// Async global->LDS copy (GLOBAL_LOAD_ASYNC_TO_LDS_B128, GVS mode):
//   LDS[ldsoff] = MEM[saddr + goff], 16 bytes per lane, tracked by ASYNCcnt.
__device__ __forceinline__ void async_copy_b128(const void* gbase, unsigned goff,
                                                unsigned ldsoff) {
    asm volatile("global_load_async_to_lds_b128 %0, %1, %2"
                 :: "v"(ldsoff), "v"(goff), "s"(gbase) : "memory");
}
__device__ __forceinline__ void wait_async() {
    asm volatile("s_wait_asynccnt 0x0" ::: "memory");
}
__device__ __forceinline__ unsigned lds_off(const void* p) {
    return (unsigned)(size_t)p;          // addrspace(3) ptr value == LDS byte offset
}

// ---------------------------------------------------------------------------
// fp32 -> bf16 conversion (grid-stride)
// ---------------------------------------------------------------------------
__global__ void cvt_f32_bf16(const float* __restrict__ in,
                             unsigned short* __restrict__ out, int n) {
    int i = blockIdx.x * blockDim.x + threadIdx.x;
    int stride = gridDim.x * blockDim.x;
    for (; i < n; i += stride) out[i] = f2bf(in[i]);
}

// fp32 [K,N] -> bf16 transposed [N,K]  (one-shot weight prep; reads coalesced)
__global__ void cvt_tr_f32_bf16(const float* __restrict__ in,
                                unsigned short* __restrict__ out, int K, int N) {
    int i = blockIdx.x * blockDim.x + threadIdx.x;
    int stride = gridDim.x * blockDim.x;
    int total = K * N;
    for (; i < total; i += stride) {
        int k = i / N, n = i - k * N;
        out[(size_t)n * K + k] = f2bf(in[i]);
    }
}

// ---------------------------------------------------------------------------
// bf16 WMMA GEMM: C[M,N] = A[M,K] * BT[N,K]^T (+bias, fused epilogue)
//   mode 0: out bf16 head-major [b,h,s,64],  val=(acc+bias)*scaleOut  (Q/K)
//   mode 1: out bf16 row-major [M,N],        val=gelu(acc+bias)       (FFN1)
//   mode 2: out f32  row-major [M,N],        val=acc+bias             (FFN2)
//   mode 3: out bf16 transposed head [b,h,d,S], val=acc+bias          (V)
// Macro-tile 128x128, 8 waves; wave = 32x64 sub-tile (2 m x 4 n tiles)
// -> 16 WMMA per 64-K step. Both tiles staged by async global->LDS with
// LDS double-buffering: copies for step i+1 overlap WMMA of step i.
// ---------------------------------------------------------------------------
__global__ __launch_bounds__(256)
void gemm_bf16(const unsigned short* __restrict__ A,
               const unsigned short* __restrict__ BT,
               const float* __restrict__ bias,
               void* __restrict__ out,
               int M, int N, int K, int mode, float scaleOut)
{
    __shared__ __align__(16) unsigned short As[2][128 * 80];  // [m][k], stride 80
    __shared__ __align__(16) unsigned short Bs[2][128 * 80];  // [n][k], stride 80

    const int tid  = threadIdx.x;
    const int lane = tid & 31;
    const int w    = tid >> 5;
    const int wr   = w & 3;
    const int wc   = w >> 2;
    const int mBase = blockIdx.y * 128;
    const int nBase = blockIdx.x * 128;

    const int ln   = lane & 15;          // row (A) / col (B) within 16-tile
    const int koff = (lane >> 4) * 8;    // bf16 K-offset for this half-wave
    const int hi   = lane >> 4;

    auto stage = [&](int buf, int k0) {
        #pragma unroll
        for (int i = 0; i < 4; ++i) {
            int chunk = tid + 256 * i;           // 1024 chunks of 8 bf16
            int r = chunk >> 3, c8 = (chunk & 7) * 8;
            async_copy_b128(A,  (unsigned)(((size_t)(mBase + r) * K + k0 + c8) * 2),
                            lds_off(&As[buf][r * 80 + c8]));
            async_copy_b128(BT, (unsigned)(((size_t)(nBase + r) * K + k0 + c8) * 2),
                            lds_off(&Bs[buf][r * 80 + c8]));
        }
    };

    v8f zero = {};
    v8f acc[2][4];
    #pragma unroll
    for (int mi = 0; mi < 2; ++mi)
        #pragma unroll
        for (int nt = 0; nt < 4; ++nt) acc[mi][nt] = zero;

    stage(0, 0);
    int i = 0;
    for (int k0 = 0; k0 < K; k0 += 64, ++i) {
        const int cur = i & 1;
        wait_async();
        __syncthreads();
        if (k0 + 64 < K) stage(cur ^ 1, k0 + 64);   // overlap next copy with WMMA

        #pragma unroll
        for (int c = 0; c < 2; ++c) {   // two K=32 WMMA chunks
            Frag a0, a1;
            a0.q[0] = *(const u32x4*)&As[cur][(wr * 32 + ln) * 80 + c * 32 + koff];
            a0.q[1] = *(const u32x4*)&As[cur][(wr * 32 + ln) * 80 + c * 32 + 16 + koff];
            a1.q[0] = *(const u32x4*)&As[cur][(wr * 32 + 16 + ln) * 80 + c * 32 + koff];
            a1.q[1] = *(const u32x4*)&As[cur][(wr * 32 + 16 + ln) * 80 + c * 32 + 16 + koff];
            #pragma unroll
            for (int nt = 0; nt < 4; ++nt) {
                Frag b;
                b.q[0] = *(const u32x4*)&Bs[cur][(wc * 64 + nt * 16 + ln) * 80 + c * 32 + koff];
                b.q[1] = *(const u32x4*)&Bs[cur][(wc * 64 + nt * 16 + ln) * 80 + c * 32 + 16 + koff];
                acc[0][nt] = __builtin_amdgcn_wmma_f32_16x16x32_bf16(
                    false, a0.v, false, b.v, (short)0, acc[0][nt], false, false);
                acc[1][nt] = __builtin_amdgcn_wmma_f32_16x16x32_bf16(
                    false, a1.v, false, b.v, (short)0, acc[1][nt], false, false);
            }
        }
        // no trailing barrier: next iteration's wait+barrier proves all waves
        // finished reading buffer `cur` before it is restaged.
    }

    // epilogue
    #pragma unroll
    for (int mi = 0; mi < 2; ++mi) {
        #pragma unroll
        for (int nt = 0; nt < 4; ++nt) {
            #pragma unroll
            for (int v = 0; v < 8; ++v) {
                int row = mBase + wr * 32 + mi * 16 + v + 8 * hi;
                int col = nBase + wc * 64 + nt * 16 + ln;
                float val = acc[mi][nt][v] + bias[col];
                int b = row >> 12, s = row & (S_ - 1);
                int h = col >> 6,  dd = col & 63;
                if (mode == 0) {
                    val *= scaleOut;
                    ((unsigned short*)out)[(((size_t)(b * H_ + h)) * S_ + s) * 64 + dd] =
                        f2bf(val);
                } else if (mode == 1) {
                    val = 0.5f * val * (1.0f + erff(val * 0.70710678118654752f));
                    ((unsigned short*)out)[(size_t)row * N + col] = f2bf(val);
                } else if (mode == 2) {
                    ((float*)out)[(size_t)row * N + col] = val;
                } else { // mode 3: V in transposed head layout [b,h,d,S]
                    ((unsigned short*)out)[(((size_t)(b * H_ + h)) * 64 + dd) * S_ + s] =
                        f2bf(val);
                }
            }
        }
    }
}

// ---------------------------------------------------------------------------
// BigBird streaming attention: one block per (b, h, qblock).
// 4 waves; wave w owns queries [w*16, w*16+16). Online softmax over the
// key-block list. All tiles (Q, K row-major, V pre-transposed [b,h,d,S])
// staged with async global->LDS; K/V double-buffered so the copy of block
// it+1 overlaps the WMMA + softmax of block it.
// ---------------------------------------------------------------------------
__global__ __launch_bounds__(128)
void bigbird_attn(const unsigned short* __restrict__ Qh,
                  const unsigned short* __restrict__ Kh,
                  const unsigned short* __restrict__ VhT,
                  const int* __restrict__ rnd,
                  float* __restrict__ attnOut)
{
    __shared__ __align__(16) unsigned short Ks[2][64 * 80];  // [key][d]
    __shared__ __align__(16) unsigned short Vt[2][64 * 80];  // [d][key]
    __shared__ __align__(16) unsigned short Ps[64 * 80];     // Q staging, then probs
    __shared__ int list[NB];
    __shared__ int listLen;

    const int tid  = threadIdx.x;
    const int lane = tid & 31;
    const int w    = tid >> 5;
    const int flat = blockIdx.x;
    const int qb   = flat & (NB - 1);
    const int h    = (flat >> 6) % H_;
    const int b    = flat / (NB * H_);
    const size_t headBase = ((size_t)(b * H_ + h)) * S_ * 64;   // Q/K [b,h,s,d]
    const size_t vhead    = ((size_t)(b * H_ + h)) * 64 * S_;   // V  [b,h,d,s]

    if (tid == 0) {
        int n = 0;
        if (qb == 0 || qb == NB - 1) {
            for (int i = 0; i < NB; ++i) list[n++] = i;          // full attention rows
        } else if (qb == 1) {
            list[n++] = 0; list[n++] = 1; list[n++] = 2; list[n++] = NB - 1;
            for (int j = 0; j < 3; ++j) list[n++] = rnd[h * (NB - 2) * 3 + 0 * 3 + j];
        } else if (qb == NB - 2) {
            list[n++] = 0; list[n++] = NB - 3; list[n++] = NB - 2; list[n++] = NB - 1;
            for (int j = 0; j < 3; ++j) list[n++] = rnd[h * (NB - 2) * 3 + (NB - 3) * 3 + j];
        } else {
            list[n++] = 0; list[n++] = qb - 1; list[n++] = qb; list[n++] = qb + 1;
            for (int j = 0; j < 3; ++j) list[n++] = rnd[h * (NB - 2) * 3 + (qb - 1) * 3 + j];
            list[n++] = NB - 1;
        }
        listLen = n;
    }
    __syncthreads();                       // list/listLen visible before staging
    const int LL = listLen;

    auto stageKV = [&](int buf, int kb) {
        const size_t kBase = headBase + (size_t)(kb * 64) * 64;
        #pragma unroll
        for (int i = 0; i < 4; ++i) {
            int chunk = tid + 128 * i;     // 512 chunks of 8 bf16 each
            int r = chunk >> 3, c8 = (chunk & 7) * 8;
            async_copy_b128(Kh, (unsigned)((kBase + (size_t)r * 64 + c8) * 2),
                            lds_off(&Ks[buf][r * 80 + c8]));
            async_copy_b128(VhT,
                            (unsigned)((vhead + (size_t)r * S_ + (size_t)(kb * 64) + c8) * 2),
                            lds_off(&Vt[buf][r * 80 + c8]));
        }
    };

    // prologue: Q -> Ps, first K/V block -> buffer 0 (single async batch)
    #pragma unroll
    for (int i = 0; i < 4; ++i) {
        int chunk = tid + 128 * i;
        int r = chunk >> 3, c8 = (chunk & 7) * 8;
        async_copy_b128(Qh, (unsigned)((headBase + (size_t)(qb * 64 + r) * 64 + c8) * 2),
                        lds_off(&Ps[r * 80 + c8]));
    }
    stageKV(0, list[0]);
    wait_async();
    __syncthreads();

    const int ln   = lane & 15;
    const int koff = (lane >> 4) * 8;
    Frag qa[2];
    #pragma unroll
    for (int c = 0; c < 2; ++c) {
        qa[c].q[0] = *(const u32x4*)&Ps[(w * 16 + ln) * 80 + c * 32 + koff];
        qa[c].q[1] = *(const u32x4*)&Ps[(w * 16 + ln) * 80 + c * 32 + 16 + koff];
    }

    v8f zero = {};
    v8f o[4] = {zero, zero, zero, zero};
    float mst[8], lst[8];
    #pragma unroll
    for (int v = 0; v < 8; ++v) { mst[v] = -1e30f; lst[v] = 0.f; }

    for (int it = 0; it < LL; ++it) {
        const int cur = it & 1;
        wait_async();
        __syncthreads();
        if (it + 1 < LL) stageKV(cur ^ 1, list[it + 1]);  // overlap copy with math
        if (it + 2 < LL) {                                // warm L2 for block after next
            const size_t nk = headBase + (size_t)(list[it + 2] * 64) * 64;
            __builtin_prefetch(&Kh[nk + (size_t)tid * 32], 0, 1);
            __builtin_prefetch(&VhT[vhead + (size_t)(tid >> 1) * S_ +
                                    (size_t)(list[it + 2] * 64) + (tid & 1) * 32], 0, 1);
        }

        // scores S = Q * K^T  (4 key-subtiles x 2 K-chunks)
        v8f sc[4];
        #pragma unroll
        for (int nt = 0; nt < 4; ++nt) {
            v8f s = zero;
            #pragma unroll
            for (int c = 0; c < 2; ++c) {
                Frag kf;
                kf.q[0] = *(const u32x4*)&Ks[cur][(nt * 16 + ln) * 80 + c * 32 + koff];
                kf.q[1] = *(const u32x4*)&Ks[cur][(nt * 16 + ln) * 80 + c * 32 + 16 + koff];
                s = __builtin_amdgcn_wmma_f32_16x16x32_bf16(false, qa[c].v, false, kf.v,
                                                            (short)0, s, false, false);
            }
            sc[nt] = s;
        }

        // online softmax: row m = v + 8*(lane>>4); reduce across 16-lane halves
        #pragma unroll
        for (int v = 0; v < 8; ++v) {
            float mx = fmaxf(fmaxf(sc[0][v], sc[1][v]), fmaxf(sc[2][v], sc[3][v]));
            #pragma unroll
            for (int msk = 1; msk < 16; msk <<= 1) mx = fmaxf(mx, __shfl_xor(mx, msk, 32));
            float mnew  = fmaxf(mst[v], mx);
            float alpha = __expf(mst[v] - mnew);
            mst[v] = mnew;
            o[0][v] *= alpha; o[1][v] *= alpha; o[2][v] *= alpha; o[3][v] *= alpha;
            float ps = 0.f;
            #pragma unroll
            for (int nt = 0; nt < 4; ++nt) {
                float p = __expf(sc[nt][v] - mnew);
                sc[nt][v] = p;
                ps += p;
            }
            #pragma unroll
            for (int msk = 1; msk < 16; msk <<= 1) ps += __shfl_xor(ps, msk, 32);
            lst[v] = lst[v] * alpha + ps;
        }

        // P -> LDS (per-wave region), re-layout for A operand
        #pragma unroll
        for (int nt = 0; nt < 4; ++nt)
            #pragma unroll
            for (int v = 0; v < 8; ++v)
                Ps[(w * 16 + v + 8 * (lane >> 4)) * 80 + nt * 16 + ln] = f2bf(sc[nt][v]);

        // O += P * V   (per-wave LDS RAW: DS ops in-order within wave)
        #pragma unroll
        for (int c = 0; c < 2; ++c) {
            Frag pa;
            pa.q[0] = *(const u32x4*)&Ps[(w * 16 + ln) * 80 + c * 32 + koff];
            pa.q[1] = *(const u32x4*)&Ps[(w * 16 + ln) * 80 + c * 32 + 16 + koff];
            #pragma unroll
            for (int nt = 0; nt < 4; ++nt) {
                Frag bv;
                bv.q[0] = *(const u32x4*)&Vt[cur][(nt * 16 + ln) * 80 + c * 32 + koff];
                bv.q[1] = *(const u32x4*)&Vt[cur][(nt * 16 + ln) * 80 + c * 32 + 16 + koff];
                o[nt] = __builtin_amdgcn_wmma_f32_16x16x32_bf16(false, pa.v, false, bv.v,
                                                                (short)0, o[nt], false, false);
            }
        }
        // no trailing barrier: next iteration's wait+barrier protects restage of `cur`.
    }

    // normalize and write context back in model layout [b, s, h*64+d]
    #pragma unroll
    for (int v = 0; v < 8; ++v) {
        float rinv = 1.0f / lst[v];
        int srow = qb * 64 + w * 16 + v + 8 * (lane >> 4);
        size_t base = ((size_t)b * S_ + srow) * DM + h * 64;
        #pragma unroll
        for (int nt = 0; nt < 4; ++nt)
            attnOut[base + nt * 16 + ln] = o[nt][v] * rinv;
    }
}

// ---------------------------------------------------------------------------
// residual + LayerNorm; optionally also emit bf16 copy for next GEMM
// ---------------------------------------------------------------------------
__global__ __launch_bounds__(256)
void residual_ln(const float* __restrict__ a, const float* __restrict__ r,
                 const float* __restrict__ g, const float* __restrict__ be,
                 float* __restrict__ outF, unsigned short* __restrict__ outB)
{
    __shared__ float red[256];
    const int row = blockIdx.x, t = threadIdx.x;
    float x[3];
    float s = 0.f;
    #pragma unroll
    for (int j = 0; j < 3; ++j) {
        int i = t + 256 * j;
        x[j] = a[(size_t)row * DM + i] + r[(size_t)row * DM + i];
        s += x[j];
    }
    red[t] = s;
    __syncthreads();
    for (int off = 128; off > 0; off >>= 1) {
        if (t < off) red[t] += red[t + off];
        __syncthreads();
    }
    float mu = red[0] / (float)DM;
    __syncthreads();
    float s2 = 0.f;
    #pragma unroll
    for (int j = 0; j < 3; ++j) { float d = x[j] - mu; s2 += d * d; }
    red[t] = s2;
    __syncthreads();
    for (int off = 128; off > 0; off >>= 1) {
        if (t < off) red[t] += red[t + off];
        __syncthreads();
    }
    float rs = rsqrtf(red[0] / (float)DM + 1e-5f);
    #pragma unroll
    for (int j = 0; j < 3; ++j) {
        int i = t + 256 * j;
        float y = (x[j] - mu) * rs * g[i] + be[i];
        outF[(size_t)row * DM + i] = y;
        if (outB) outB[(size_t)row * DM + i] = f2bf(y);
    }
}

// ---------------------------------------------------------------------------
extern "C" void kernel_launch(void* const* d_in, const int* in_sizes, int n_in,
                              void* d_out, int out_size, void* d_ws, size_t ws_size,
                              hipStream_t stream)
{
    const float* x   = (const float*)d_in[0];
    const int*   rnd = (const int*)  d_in[1];
    const float* Wq  = (const float*)d_in[2];  const float* bq  = (const float*)d_in[3];
    const float* Wk  = (const float*)d_in[4];  const float* bk  = (const float*)d_in[5];
    const float* Wv  = (const float*)d_in[6];  const float* bv  = (const float*)d_in[7];
    const float* g1  = (const float*)d_in[8];  const float* be1 = (const float*)d_in[9];
    const float* W1  = (const float*)d_in[10]; const float* b1  = (const float*)d_in[11];
    const float* W2  = (const float*)d_in[12]; const float* b2  = (const float*)d_in[13];
    const float* g2  = (const float*)d_in[14]; const float* be2 = (const float*)d_in[15];

    char* ws = (char*)d_ws;
    size_t off = 0;
    auto alloc = [&](size_t bytes) {
        void* p = ws + off;
        off += (bytes + 255) & ~(size_t)255;
        return p;
    };
    unsigned short* xb  = (unsigned short*)alloc((size_t)M_ * DM * 2);
    unsigned short* Wqt = (unsigned short*)alloc((size_t)DM * DM * 2);   // [N,K]
    unsigned short* Wkt = (unsigned short*)alloc((size_t)DM * DM * 2);
    unsigned short* Wvt = (unsigned short*)alloc((size_t)DM * DM * 2);
    unsigned short* Qh  = (unsigned short*)alloc((size_t)M_ * DM * 2);   // [b,h,s,d]
    unsigned short* Kh  = (unsigned short*)alloc((size_t)M_ * DM * 2);   // [b,h,s,d]
    unsigned short* VhT = (unsigned short*)alloc((size_t)M_ * DM * 2);   // [b,h,d,s]
    float*          at  = (float*)         alloc((size_t)M_ * DM * 4);
    float*          hF  = (float*)         alloc((size_t)M_ * DM * 4);
    unsigned short* hB  = (unsigned short*)alloc((size_t)M_ * DM * 2);
    unsigned short* W1t = (unsigned short*)alloc((size_t)DM * FF * 2);   // [FF,DM]
    unsigned short* W2t = (unsigned short*)alloc((size_t)FF * DM * 2);   // [DM,FF]
    unsigned short* f1  = (unsigned short*)alloc((size_t)M_ * FF * 2);
    float*          f2  = (float*)         alloc((size_t)M_ * DM * 4);

    // 1) fp32 -> bf16 staging (weights pre-transposed to [N,K] for the
    //    B operand so every GEMM tile is a contiguous async copy)
    cvt_f32_bf16<<<512, 256, 0, stream>>>(x,  xb,  M_ * DM);
    cvt_tr_f32_bf16<<<256, 256, 0, stream>>>(Wq, Wqt, DM, DM);
    cvt_tr_f32_bf16<<<256, 256, 0, stream>>>(Wk, Wkt, DM, DM);
    cvt_tr_f32_bf16<<<256, 256, 0, stream>>>(Wv, Wvt, DM, DM);
    cvt_tr_f32_bf16<<<512, 256, 0, stream>>>(W1, W1t, DM, FF);
    cvt_tr_f32_bf16<<<512, 256, 0, stream>>>(W2, W2t, FF, DM);

    // 2) QKV projections (Q pre-scaled by 1/sqrt(64); V written transposed)
    dim3 gQKV(DM / 128, M_ / 128);
    gemm_bf16<<<gQKV, 256, 0, stream>>>(xb, Wqt, bq, Qh,  M_, DM, DM, 0, 0.125f);
    gemm_bf16<<<gQKV, 256, 0, stream>>>(xb, Wkt, bk, Kh,  M_, DM, DM, 0, 1.0f);
    gemm_bf16<<<gQKV, 256, 0, stream>>>(xb, Wvt, bv, VhT, M_, DM, DM, 3, 1.0f);

    // 3) BigBird sparse attention
    bigbird_attn<<<B_ * H_ * NB, 128, 0, stream>>>(Qh, Kh, VhT, rnd, at);

    // 4) LN1(attn + x) -> hF (f32) + hB (bf16)
    residual_ln<<<M_, 256, 0, stream>>>(at, x, g1, be1, hF, hB);

    // 5) FFN: gelu(h@W1+b1) @ W2 + b2
    dim3 gF1(FF / 128, M_ / 128);
    gemm_bf16<<<gF1, 256, 0, stream>>>(hB, W1t, b1, f1, M_, FF, DM, 1, 1.0f);
    dim3 gF2(DM / 128, M_ / 128);
    gemm_bf16<<<gF2, 256, 0, stream>>>(f1, W2t, b2, f2, M_, DM, FF, 2, 1.0f);

    // 6) LN2(ffn + h) -> d_out (f32)
    residual_ln<<<M_, 256, 0, stream>>>(f2, hF, g2, be2, (float*)d_out, nullptr);
}
